// ChiralRetro_25924422599320
// MI455X (gfx1250) — matrix operand order repaired and checked by reading, not compiled, not run on hardware
//
// ChiralRetro pipeline for MI455X (gfx1250): DMPNN encoder + MHA + readout heads.
// GEMMs use v_wmma_f32_16x16x32_bf16 (fp32 accumulate); gathers/bias/ReLU fused.
// B-tile staging: global_load_async_to_lds_b128 (ASYNCcnt). A-tile staging:
// b128 fp32 loads -> bf16 pack -> one ds_store_b128 (no guards; K pre-padded).
#include <hip/hip_runtime.h>
#include <hip/hip_bf16.h>
#include <stdint.h>

typedef __attribute__((ext_vector_type(16))) __bf16 v16bf;
typedef __attribute__((ext_vector_type(8)))  float  v8f;
typedef __attribute__((ext_vector_type(4)))  int    v4i;

#define NB_MOL   512
#define NA_PER   64
#define NEU_PER  128
#define NATOMS   32768      // 512*64
#define NDIR     131072     // 512*256
#define NUND     65536      // 512*128
#define FA       98
#define FB       111
#define HD       256
#define NHEADS   8
#define DKH      32
#define NMLP     512
#define NAOUT    128
#define NBOUT    32
#define OUT_PM   12289      // 128*32 + 64*128 + 1

#if defined(__has_builtin)
#if __has_builtin(__builtin_amdgcn_global_load_async_to_lds_b128)
#define HAVE_ASYNC_LDS 1
#endif
#endif
#ifndef HAVE_ASYNC_LDS
#define HAVE_ASYNC_LDS 0
#endif

#define LDS_AS __attribute__((address_space(3)))
#define GLB_AS __attribute__((address_space(1)))

union Frag16 { v16bf v; uint4 u[2]; __bf16 e[16]; };

// ------- stage 8 consecutive A elements -> bf16 LDS (one ds_store_b128) ----
// Exact paths only: all call sites guarantee lda%4==0, k0%8==0, Kreal==Kpad.
// 0 PLAIN: A0[row][k]   1 MSG: A0[b2a[r]][k]-A1[b2rev[r]][k]
// 3 PAIRS: k<k0 ? A0[pairs[2r]][k] : A0[pairs[2r+1]][k-k0]
template<int AMODE>
__device__ __forceinline__ void stage8(
    __bf16* __restrict__ dst,
    const float* __restrict__ A0, const float* __restrict__ A1,
    const int* __restrict__ idx0, const int* __restrict__ idx1,
    int lda0, int k0, int row, int kg)
{
  union { __bf16 e[8]; uint4 u; } t;
  if (AMODE == 1) {
    const float4* pa = (const float4*)(A0 + (size_t)idx0[row] * lda0 + kg);
    const float4* pb = (const float4*)(A1 + (size_t)idx1[row] * lda0 + kg);
    float4 a0 = pa[0], a1 = pa[1], b0 = pb[0], b1 = pb[1];
    t.e[0] = (__bf16)(a0.x - b0.x); t.e[1] = (__bf16)(a0.y - b0.y);
    t.e[2] = (__bf16)(a0.z - b0.z); t.e[3] = (__bf16)(a0.w - b0.w);
    t.e[4] = (__bf16)(a1.x - b1.x); t.e[5] = (__bf16)(a1.y - b1.y);
    t.e[6] = (__bf16)(a1.z - b1.z); t.e[7] = (__bf16)(a1.w - b1.w);
  } else {
    const float* src;
    if (AMODE == 0) {
      src = A0 + (size_t)row * lda0 + kg;
    } else {                          // PAIRS
      bool lo = (kg < k0);
      int p = idx0[2 * row + (lo ? 0 : 1)];
      src = A0 + (size_t)p * lda0 + (lo ? kg : kg - k0);
    }
    float4 x0 = ((const float4*)src)[0], x1 = ((const float4*)src)[1];
    t.e[0] = (__bf16)x0.x; t.e[1] = (__bf16)x0.y;
    t.e[2] = (__bf16)x0.z; t.e[3] = (__bf16)x0.w;
    t.e[4] = (__bf16)x1.x; t.e[5] = (__bf16)x1.y;
    t.e[6] = (__bf16)x1.z; t.e[7] = (__bf16)x1.w;
  }
  *(uint4*)dst = t.u;
}

// ---------------- WMMA GEMM core: C[M,N] = epi(A[M,K] @ B[K,N]) ------------
// Bt is bf16, LDS-ready layout [N][Kpad] (B^T row-major), zero-padded.
// Wave tile 32x32 (4 accumulators), block tile (WM*32)x(WN*32), 8 waves.
// M % BM == 0 and N % BN == 0 at every call site (no epilogue guards).
template<int WM, int WN, int AMODE>
__device__ __forceinline__ void gemm_impl(
    const float* __restrict__ A0, const float* __restrict__ A1,
    const int* __restrict__ idx0, const int* __restrict__ idx1,
    int lda0, int k0,
    const __bf16* __restrict__ Bt, int Kpad,
    float* __restrict__ C,
    const float* __restrict__ addsrc, const float* __restrict__ bias,
    int do_relu, int N)
{
  constexpr int BM = WM * 32, BN = WN * 32, NT = WM * WN * 32;
  constexpr int LDT = 40;                       // row stride (bf16); 80B = 16B-aligned
  __shared__ __bf16 As[BM * LDT];
  __shared__ __bf16 Bs[BN * LDT];

  const int tid  = threadIdx.x;
  const int lane = tid & 31, wv = tid >> 5;
  const int wm = wv / WN, wn = wv % WN;
  const int l15 = lane & 15, hf = lane >> 4;
  const int mblk = blockIdx.x * BM, nblk = blockIdx.y * BN;

  v8f acc[2][2] = {};
  const int ktiles = Kpad >> 5;

  for (int kt = 0; kt < ktiles; ++kt) {
    __syncthreads();
    // --- stage B tile: bf16 [N][Kpad] -> LDS [n][k] (async, ASYNCcnt) ---
#if HAVE_ASYNC_LDS
    for (int i = tid; i < BN * 2; i += NT) {
      int n = i >> 1, kb = (i & 1) * 16;
      const __bf16* src = Bt + (size_t)(nblk + n) * Kpad + kt * 32 + kb;
      __builtin_amdgcn_global_load_async_to_lds_b128(
          (GLB_AS v4i*)src, (LDS_AS v4i*)&Bs[n * LDT + kb], 0, 0);
    }
#else
    for (int i = tid; i < BN * 2; i += NT) {
      int n = i >> 1, kb = (i & 1) * 16;
      const __bf16* src = Bt + (size_t)(nblk + n) * Kpad + kt * 32 + kb;
      *(uint4*)&Bs[n * LDT + kb] = *(const uint4*)src;
    }
#endif
    // --- stage A tile: fp32 global (fused gather) -> bf16 LDS [m][k] ---
    for (int i = tid; i < BM * 4; i += NT) {
      int r = i >> 2, kb = (i & 3) * 8;
      stage8<AMODE>(&As[r * LDT + kb], A0, A1, idx0, idx1,
                    lda0, k0, mblk + r, kt * 32 + kb);
    }
#if HAVE_ASYNC_LDS
    asm volatile("s_wait_asynccnt 0x0" ::: "memory");
#endif
    __syncthreads();

    // --- fragments per ISA 7.12.2 (16-bit A 16x32, B as [n][k]) ---
    Frag16 af[2], bfg[2];
#pragma unroll
    for (int mi = 0; mi < 2; ++mi) {
      const __bf16* p = &As[(wm * 32 + mi * 16 + l15) * LDT + hf * 8];
      af[mi].u[0] = *(const uint4*)p;
      af[mi].u[1] = *(const uint4*)(p + 16);
    }
#pragma unroll
    for (int ni = 0; ni < 2; ++ni) {
      const __bf16* p = &Bs[(wn * 32 + ni * 16 + l15) * LDT + hf * 8];
      bfg[ni].u[0] = *(const uint4*)p;
      bfg[ni].u[1] = *(const uint4*)(p + 16);
    }
#pragma unroll
    for (int mi = 0; mi < 2; ++mi)
#pragma unroll
      for (int ni = 0; ni < 2; ++ni)
        acc[mi][ni] = __builtin_amdgcn_wmma_f32_16x16x32_bf16(
            false, af[mi].v, false, bfg[ni].v, (short)0, acc[mi][ni], false, false);
  }

  // --- fused epilogue: +addsrc, +bias, relu (no bounds guards needed) ---
#pragma unroll
  for (int mi = 0; mi < 2; ++mi)
#pragma unroll
    for (int ni = 0; ni < 2; ++ni)
#pragma unroll
      for (int r = 0; r < 8; ++r) {
        int grow = mblk + wm * 32 + mi * 16 + hf * 8 + r;
        int gcol = nblk + wn * 32 + ni * 16 + l15;
        float val = acc[mi][ni][r];
        if (addsrc) val += addsrc[(size_t)grow * N + gcol];
        if (bias)   val += bias[gcol];
        if (do_relu) val = fmaxf(val, 0.f);
        C[(size_t)grow * N + gcol] = val;
      }
}

#define GEMM_WRAPPER(name, WM, WN, AM)                                        \
__global__ __launch_bounds__(256, 1)                                          \
void name(const float* A0, const float* A1, const int* i0, const int* i1,     \
          int lda0, int k0, const __bf16* Bt, int Kpad, float* C,             \
          const float* add, const float* bias, int relu, int N)               \
{ gemm_impl<WM,WN,AM>(A0,A1,i0,i1,lda0,k0,Bt,Kpad,C,add,bias,relu,N); }

GEMM_WRAPPER(k_gemm_plain, 2, 4, 0)
GEMM_WRAPPER(k_gemm_msg,   2, 4, 1)
GEMM_WRAPPER(k_gemm_pairs, 2, 4, 3)
GEMM_WRAPPER(k_gemm_n32,   8, 1, 0)

// ------------- weight convert: fp32 W -> bf16 B^T [N][Kpad], zero-pad ------
__global__ void k_wconv(const float* __restrict__ W, __bf16* __restrict__ Bt,
                        int K, int N, int Kpad, int trans)
{
  size_t i = (size_t)blockIdx.x * blockDim.x + threadIdx.x;
  if (i >= (size_t)N * Kpad) return;
  int n = (int)(i / Kpad), kp = (int)(i % Kpad);
  float v = 0.f;
  if (kp < K) v = trans ? W[(size_t)kp * N + n] : W[(size_t)n * K + kp];
  Bt[i] = (__bf16)v;
}

// ------- row-pack: dst[r][0:K0]=src0, [K0:K0+K1]=src1, rest zero -----------
__global__ void k_cat2(const float* __restrict__ s0, int K0,
                       const float* __restrict__ s1, int K1,
                       float* __restrict__ dst, int Kpad, int nrows)
{
  size_t i = (size_t)blockIdx.x * blockDim.x + threadIdx.x;
  if (i >= (size_t)nrows * Kpad) return;
  int r = (int)(i / Kpad), k = (int)(i % Kpad);
  float v = 0.f;
  if (k < K0) v = s0[(size_t)r * K0 + k];
  else if (k < K0 + K1) v = s1[(size_t)r * K1 + (k - K0)];
  dst[i] = v;
}

__global__ void k_zero(float* p, size_t n) {
  size_t i = (size_t)blockIdx.x * blockDim.x + threadIdx.x;
  size_t st = (size_t)gridDim.x * blockDim.x;
  for (; i < n; i += st) p[i] = 0.f;
}

// ------------- segment sum: a_sum[b2t[e]] += h[e] (fp32 atomics) -----------
__global__ void k_segsum(const float* __restrict__ h, const int* __restrict__ b2t,
                         float* __restrict__ asum)
{
  int e = blockIdx.x;                  // uniform per block -> scalar load
  int t = b2t[e];
  int c = threadIdx.x;                 // 256 = HD
  atomicAdd(&asum[(size_t)t * HD + c], h[(size_t)e * HD + c]);
}

// ------------- fused per-(molecule,head) attention: S, softmax, P@V --------
__global__ __launch_bounds__(32, 1)
void k_attn(const float* __restrict__ q, const float* __restrict__ k,
            const float* __restrict__ v, float* __restrict__ ctx)
{
  const int b = blockIdx.x, hd = blockIdx.y;
  const int lane = threadIdx.x;
  const int l15 = lane & 15, hf = lane >> 4;
  const size_t base = (size_t)b * NA_PER * HD + (size_t)hd * DKH;

  __shared__ __bf16 vT[DKH * 72];      // [dk][atom]  (B^T for P@V)
  __shared__ __bf16 P [NA_PER * 72];   // [q_atom][k_atom] probabilities

  for (int a = 0; a < NA_PER; ++a)     // lane == dk
    vT[lane * 72 + a] = (__bf16)v[base + (size_t)a * HD + lane];

  // S = q @ k^T : 4x4 WMMA tiles, K=32 (one 16x16x32 each)
  v8f acc[4][4] = {};
  Frag16 af[4], bfg[4];
#pragma unroll
  for (int mi = 0; mi < 4; ++mi) {
    const float* p = q + base + (size_t)(mi * 16 + l15) * HD + hf * 8;
#pragma unroll
    for (int j = 0; j < 8; ++j) { af[mi].e[j] = (__bf16)p[j]; af[mi].e[8 + j] = (__bf16)p[16 + j]; }
  }
#pragma unroll
  for (int ni = 0; ni < 4; ++ni) {
    const float* p = k + base + (size_t)(ni * 16 + l15) * HD + hf * 8;
#pragma unroll
    for (int j = 0; j < 8; ++j) { bfg[ni].e[j] = (__bf16)p[j]; bfg[ni].e[8 + j] = (__bf16)p[16 + j]; }
  }
#pragma unroll
  for (int mi = 0; mi < 4; ++mi)
#pragma unroll
    for (int ni = 0; ni < 4; ++ni)
      acc[mi][ni] = __builtin_amdgcn_wmma_f32_16x16x32_bf16(
          false, af[mi].v, false, bfg[ni].v, (short)0, acc[mi][ni], false, false);

  // in-register softmax over rows; 16-lane shfl_xor reductions (wave32 halves)
  const float scale = 0.1767766952966369f;   // 1/sqrt(32)
#pragma unroll
  for (int mi = 0; mi < 4; ++mi)
#pragma unroll
    for (int r = 0; r < 8; ++r) {
      float mx = acc[mi][0][r];
#pragma unroll
      for (int ni = 1; ni < 4; ++ni) mx = fmaxf(mx, acc[mi][ni][r]);
#pragma unroll
      for (int off = 8; off >= 1; off >>= 1) mx = fmaxf(mx, __shfl_xor(mx, off, 16));
      float pv[4]; float sum = 0.f;
#pragma unroll
      for (int ni = 0; ni < 4; ++ni) { pv[ni] = __expf((acc[mi][ni][r] - mx) * scale); sum += pv[ni]; }
#pragma unroll
      for (int off = 8; off >= 1; off >>= 1) sum += __shfl_xor(sum, off, 16);
      float inv = 1.f / sum;
      int m = mi * 16 + hf * 8 + r;
#pragma unroll
      for (int ni = 0; ni < 4; ++ni) P[m * 72 + ni * 16 + l15] = (__bf16)(pv[ni] * inv);
    }
  __syncthreads();

  // ctx = P @ V : per 16-row block, N=32 (2 tiles), K=64 (2 WMMA)
#pragma unroll
  for (int mi = 0; mi < 4; ++mi) {
    v8f cacc[2] = {};
#pragma unroll
    for (int kt = 0; kt < 2; ++kt) {
      Frag16 pa;
      const __bf16* pp = &P[(mi * 16 + l15) * 72 + kt * 32 + hf * 8];
      pa.u[0] = *(const uint4*)pp;
      pa.u[1] = *(const uint4*)(pp + 16);
#pragma unroll
      for (int ni = 0; ni < 2; ++ni) {
        Frag16 vb;
        const __bf16* vp = &vT[(ni * 16 + l15) * 72 + kt * 32 + hf * 8];
        vb.u[0] = *(const uint4*)vp;
        vb.u[1] = *(const uint4*)(vp + 16);
        cacc[ni] = __builtin_amdgcn_wmma_f32_16x16x32_bf16(
            false, pa.v, false, vb.v, (short)0, cacc[ni], false, false);
      }
    }
#pragma unroll
    for (int ni = 0; ni < 2; ++ni)
#pragma unroll
      for (int r = 0; r < 8; ++r) {
        int atom = mi * 16 + hf * 8 + r, dk = ni * 16 + l15;
        ctx[base + (size_t)atom * HD + dk] = cacc[ni][r];
      }
  }
}

// ------------- graph pooling / tiny head / output assembly -----------------
__global__ void k_gsum(const float* __restrict__ af, float* __restrict__ gv) {
  int b = blockIdx.x, c = threadIdx.x;        // 256 cols
  float s = 0.f;
  for (int a = 0; a < NA_PER; ++a) s += af[((size_t)b * NA_PER + a) * HD + c];
  gv[(size_t)b * HD + c] = s;
}

__global__ void k_g2(const float* __restrict__ gh, const float* __restrict__ G2w,
                     const float* __restrict__ g2s, float* __restrict__ gout) {
  int b = blockIdx.x * blockDim.x + threadIdx.x;
  if (b >= NB_MOL) return;
  float s = g2s[0];
  for (int j = 0; j < NMLP; ++j) s += gh[(size_t)b * NMLP + j] * G2w[j];
  gout[b] = s;
}

__global__ void k_assemble(const float* __restrict__ bo, const float* __restrict__ ao,
                           const float* __restrict__ go, float* __restrict__ out) {
  size_t i = (size_t)blockIdx.x * blockDim.x + threadIdx.x;
  if (i >= (size_t)NB_MOL * OUT_PM) return;
  int mol = (int)(i / OUT_PM), j = (int)(i % OUT_PM);
  float v;
  if (j < NEU_PER * NBOUT) {
    v = bo[((size_t)mol * NEU_PER + (j >> 5)) * NBOUT + (j & 31)];
  } else if (j < NEU_PER * NBOUT + NA_PER * NAOUT) {
    int jj = j - NEU_PER * NBOUT;
    v = ao[((size_t)mol * NA_PER + (jj >> 7)) * NAOUT + (jj & 127)];
  } else v = go[mol];
  out[i] = v;
}

// ===========================================================================
extern "C" void kernel_launch(void* const* d_in, const int* in_sizes, int n_in,
                              void* d_out, int out_size, void* d_ws, size_t ws_size,
                              hipStream_t stream)
{
  (void)in_sizes; (void)n_in; (void)out_size; (void)ws_size;
  const float* f_atoms  = (const float*)d_in[0];
  const float* f_bonds  = (const float*)d_in[1];
  const float* prev_hid = (const float*)d_in[2];
  const int*   b2a      = (const int*)d_in[3];
  const int*   b2t      = (const int*)d_in[4];
  const int*   b2rev    = (const int*)d_in[5];
  const int*   pairs    = (const int*)d_in[6];
  const float* W_i = (const float*)d_in[7],  * W_h  = (const float*)d_in[8];
  const float* W_o = (const float*)d_in[9],  * b_o  = (const float*)d_in[10];
  const float* Wq  = (const float*)d_in[11], * Wk   = (const float*)d_in[12];
  const float* Wv  = (const float*)d_in[13], * Wat  = (const float*)d_in[14];
  const float* Wvv = (const float*)d_in[15], * Wvc  = (const float*)d_in[16];
  const float* A1w = (const float*)d_in[17], * a1   = (const float*)d_in[18];
  const float* A2w = (const float*)d_in[19], * a2   = (const float*)d_in[20];
  const float* B1w = (const float*)d_in[21], * b1   = (const float*)d_in[22];
  const float* B2w = (const float*)d_in[23], * b2b  = (const float*)d_in[24];
  const float* G1w = (const float*)d_in[25], * g1   = (const float*)d_in[26];
  const float* G2w = (const float*)d_in[27], * g2   = (const float*)d_in[28];

  char* ws = (char*)d_ws;
  const size_t MB = 1024ull * 1024ull;
  float* h0    = (float*)(ws + 0);            // 128MB
  float* hA    = (float*)(ws + 128 * MB);     // 128MB
  float* hB    = (float*)(ws + 256 * MB);     // 128MB
  float* a_sum = (float*)(ws + 384 * MB);     // 32MB (later: a_in)
  float* a_ft  = (float*)(ws + 416 * MB);     // 32MB
  float* at_f  = (float*)(ws + 448 * MB);     // 32MB atom_feats
  float* b_out = (float*)(ws + 480 * MB);     // 8MB bond_outs
  float* gvec  = (float*)(ws + 488 * MB);
  float* ghid  = (float*)(ws + 488 * MB + 512 * 1024);
  float* gout  = (float*)(ws + 488 * MB + 1536 * 1024);
  __bf16* wb   = (__bf16*)(ws + 490 * MB);
  // region reuse:
  float* fb_pad = hA;               // [NDIR][128] fp32 (pre-encoder only)
  float* cat_in = hB;               // [NATOMS][384] fp32 (post-encoder, pre-a_ft)
  float* qb   = h0;                 float* kb   = h0 + 8388608;
  float* vb   = h0 + 16777216;      float* ctx  = h0 + 25165824;
  float* bhid = hA;                                       // 128MB
  float* af2  = hB;                 float* t1 = hB + 8388608;
  float* ahid = hB + 16777216;                            // 64MB
  float* a_out = a_sum;                                   // 16MB reuse

  // bf16 weight slots (element offsets)
  __bf16* Wi_t  = wb + 0;         __bf16* Wh_t  = wb + 32768;
  __bf16* Wo_t  = wb + 98304;     __bf16* Wq_t  = wb + 196608;
  __bf16* Wk_t  = wb + 262144;    __bf16* Wv_t  = wb + 327680;
  __bf16* Wat_t = wb + 393216;    __bf16* Wvv_p = wb + 458752;
  __bf16* Wvc_p = wb + 524288;    __bf16* A1_t  = wb + 589824;
  __bf16* A2_t  = wb + 720896;    __bf16* B1_t  = wb + 786432;
  __bf16* B2_t  = wb + 1048576;   __bf16* G1_t  = wb + 1064960;

  auto wconv = [&](const float* W, __bf16* o, int K, int N, int Kpad, int tr) {
    size_t n = (size_t)N * Kpad;
    hipLaunchKernelGGL(k_wconv, dim3((unsigned)((n + 255) / 256)), dim3(256), 0, stream,
                       W, o, K, N, Kpad, tr);
  };
  auto cat2 = [&](const float* s0, int K0, const float* s1, int K1,
                  float* dst, int Kpad, int nrows) {
    size_t n = (size_t)nrows * Kpad;
    hipLaunchKernelGGL(k_cat2, dim3((unsigned)((n + 255) / 256)), dim3(256), 0, stream,
                       s0, K0, s1, K1, dst, Kpad, nrows);
  };
  auto gemm = [&](int am, const float* Ap, const float* Aq, const int* i0, const int* i1,
                  int lda0, int k0, const __bf16* Bt, int Kp, float* C,
                  const float* add, const float* bias, int relu, int M, int N) {
    dim3 g((unsigned)(M / 64), (unsigned)(N / 128));
    if (am == 0)
      hipLaunchKernelGGL(k_gemm_plain, g, dim3(256), 0, stream, Ap, Aq, i0, i1,
                         lda0, k0, Bt, Kp, C, add, bias, relu, N);
    else if (am == 1)
      hipLaunchKernelGGL(k_gemm_msg,   g, dim3(256), 0, stream, Ap, Aq, i0, i1,
                         lda0, k0, Bt, Kp, C, add, bias, relu, N);
    else
      hipLaunchKernelGGL(k_gemm_pairs, g, dim3(256), 0, stream, Ap, Aq, i0, i1,
                         lda0, k0, Bt, Kp, C, add, bias, relu, N);
  };
  auto zero = [&](float* p, size_t n) {
    hipLaunchKernelGGL(k_zero, dim3(2048), dim3(256), 0, stream, p, n);
  };

  // ---- weight prep (tiny) ----
  wconv(W_i, Wi_t, FB, HD, 128, 1);   wconv(W_h, Wh_t, HD, HD, HD, 1);
  wconv(W_o, Wo_t, FA + HD, HD, 384, 1);
  wconv(Wq, Wq_t, HD, HD, HD, 1);     wconv(Wk, Wk_t, HD, HD, HD, 1);
  wconv(Wv, Wv_t, HD, HD, HD, 1);     wconv(Wat, Wat_t, HD, HD, HD, 1);
  wconv(Wvv, Wvv_p, HD, HD, HD, 0);   wconv(Wvc, Wvc_p, HD, HD, HD, 0);  // used as W.T
  wconv(A1w, A1_t, HD, NMLP, HD, 1);  wconv(A2w, A2_t, NMLP, NAOUT, NMLP, 1);
  wconv(B1w, B1_t, 2 * HD, NMLP, 2 * HD, 1); wconv(B2w, B2_t, NMLP, NBOUT, NMLP, 1);
  wconv(G1w, G1_t, HD, NMLP, HD, 1);

  // ---- DMPNN encoder ----
  cat2(f_bonds, FB, nullptr, 0, fb_pad, 128, NDIR);               // zero-pad K 111->128
  gemm(0, fb_pad, nullptr, nullptr, nullptr, 128, 0,
       Wi_t, 128, h0, nullptr, nullptr, 1, NDIR, HD);             // h0 = relu(f_bonds@W_i)
  float* hcur = h0;
  for (int d = 0; d < 9; ++d) {                                   // DEPTH-1
    float* hnext = (d & 1) ? hB : hA;
    zero(a_sum, (size_t)NATOMS * HD);
    hipLaunchKernelGGL(k_segsum, dim3(NDIR), dim3(HD), 0, stream, hcur, b2t, a_sum);
    gemm(1, a_sum, hcur, b2a, b2rev, HD, 0,                       // m = asum[b2a]-h[b2rev]
         Wh_t, HD, hnext, h0, nullptr, 1, NDIR, HD);              // h = relu(h0 + m@W_h)
    hcur = hnext;
  }
  zero(a_sum, (size_t)NATOMS * HD);                               // a_in
  hipLaunchKernelGGL(k_segsum, dim3(NDIR), dim3(HD), 0, stream, hcur, b2t, a_sum);
  cat2(f_atoms, FA, a_sum, HD, cat_in, 384, NATOMS);              // [f_atoms | a_in | 0]
  gemm(0, cat_in, nullptr, nullptr, nullptr, 384, 0,
       Wo_t, 384, a_ft, nullptr, b_o, 1, NATOMS, HD);

  // ---- multi-head attention ----
  gemm(0, a_ft, nullptr, nullptr, nullptr, HD, 0, Wq_t, HD, qb, nullptr, nullptr, 0, NATOMS, HD);
  gemm(0, a_ft, nullptr, nullptr, nullptr, HD, 0, Wk_t, HD, kb, nullptr, nullptr, 0, NATOMS, HD);
  gemm(0, a_ft, nullptr, nullptr, nullptr, HD, 0, Wv_t, HD, vb, nullptr, nullptr, 0, NATOMS, HD);
  hipLaunchKernelGGL(k_attn, dim3(NB_MOL, NHEADS), dim3(32), 0, stream, qb, kb, vb, ctx);
  gemm(0, ctx, nullptr, nullptr, nullptr, HD, 0,
       Wat_t, HD, af2, a_ft, nullptr, 0, NATOMS, HD);             // x + ctx@Wattn

  // ---- recurrent update: relu(prev@Wvv.T + af2@Wvc.T) ----
  gemm(0, prev_hid, nullptr, nullptr, nullptr, HD, 0, Wvv_p, HD, t1, nullptr, nullptr, 0, NATOMS, HD);
  gemm(0, af2, nullptr, nullptr, nullptr, HD, 0, Wvc_p, HD, at_f, t1, nullptr, 1, NATOMS, HD);

  // ---- heads ----
  gemm(0, at_f, nullptr, nullptr, nullptr, HD, 0, A1_t, HD, ahid, nullptr, a1, 1, NATOMS, NMLP);
  gemm(0, ahid, nullptr, nullptr, nullptr, NMLP, 0, A2_t, NMLP, a_out, nullptr, a2, 0, NATOMS, NAOUT);
  gemm(3, at_f, nullptr, pairs, nullptr, HD, HD,                  // bond-pair gather fused
       B1_t, 2 * HD, bhid, nullptr, b1, 1, NUND, NMLP);
  hipLaunchKernelGGL(k_gemm_n32, dim3(NUND / 256, 1), dim3(256), 0, stream,
       bhid, (const float*)nullptr, (const int*)nullptr, (const int*)nullptr,
       NMLP, 0, B2_t, NMLP, b_out, (const float*)nullptr, b2b, 0, NBOUT);
  hipLaunchKernelGGL(k_gsum, dim3(NB_MOL), dim3(HD), 0, stream, at_f, gvec);
  gemm(0, gvec, nullptr, nullptr, nullptr, HD, 0, G1_t, HD, ghid, nullptr, g1, 1, NB_MOL, NMLP);
  hipLaunchKernelGGL(k_g2, dim3(2), dim3(256), 0, stream, ghid, G2w, g2, gout);

  size_t tot = (size_t)NB_MOL * OUT_PM;
  hipLaunchKernelGGL(k_assemble, dim3((unsigned)((tot + 255) / 256)), dim3(256), 0, stream,
                     b_out, a_out, gout, (float*)d_out);
}
